// MaskedCrossAttention_29884382446102
// MI455X (gfx1250) — matrix operand order repaired and checked by reading, not compiled, not run on hardware
//
#include <hip/hip_runtime.h>
#include <hip/hip_bf16.h>
#include <stdint.h>

// Problem constants (match reference)
#define DIMX   1024
#define INNER  512
#define HEADS  8
#define DHEAD  64
#define BATCH  4
#define NTOK   4096
#define NLAT   1024
#define NEG_BIG (-3.0e38f)

#define HAVE_TDM __has_builtin(__builtin_amdgcn_tensor_load_to_lds)

typedef __bf16 bf16x16 __attribute__((ext_vector_type(16)));
typedef float  f32x8   __attribute__((ext_vector_type(8)));

union Frag { bf16x16 v; uint4 u[2]; };

__device__ __forceinline__ uint16_t f2bf(float f) {
  uint32_t x = __float_as_uint(f);
  x += 0x7FFFu + ((x >> 16) & 1u);   // round-to-nearest-even
  return (uint16_t)(x >> 16);
}

// LDS byte offset of a __shared__ object: flat LDS address truncates to the
// in-aperture offset (ISA 10.2: low 32 bits of the address).
__device__ __forceinline__ uint32_t lds_off(const void* p) {
  return (uint32_t)(uintptr_t)p;
}

#if HAVE_TDM
typedef unsigned int u32x4 __attribute__((ext_vector_type(4)));
typedef int          i32x4 __attribute__((ext_vector_type(4)));
typedef int          i32x8 __attribute__((ext_vector_type(8)));

// Issue one TDM 2D-tile load (bf16 elements): rows x cols tile, row stride in
// elements, into LDS at lds_byte_off. Wave-level op (EXEC ignored); caller
// must restrict to one wave per workgroup and fence with s_wait_tensorcnt +
// barrier. D# packing per CDNA5 ISA.
__device__ __forceinline__ void tdm_load_tile_2d(uint32_t lds_byte_off,
                                                 const uint16_t* gptr,
                                                 int rows, int cols,
                                                 int row_stride_elems) {
  const uint64_t ga = (uint64_t)(uintptr_t)gptr;
  const u32x4 g0 = {
      1u,                                                  // count=1 valid user D#
      lds_byte_off,                                        // lds_addr (bytes)
      (uint32_t)ga,                                        // global_addr[31:0]
      ((uint32_t)(ga >> 32) & 0x01FFFFFFu) | (2u << 30)    // global_addr[56:32] | type=2
  };
  // tensor_dim0 = tensor_dim1 = 1<<30 (tiles never OOB; dims only gate OOB)
  const i32x8 g1 = {
      0x00010000,                      // data_size=1 (2 bytes); no pad/iterate/multicast
      0,                               // abar addr=0 ; tensor_dim0[15:0]<<16 = 0
      0x4000,                          // tensor_dim0[31:16]=0x4000 ; tensor_dim1[15:0]=0
      0x4000 | (cols << 16),           // tensor_dim1[31:16]=0x4000 | tile_dim0=cols
      rows,                            // tile_dim1=rows ; tile_dim2=0
      row_stride_elems,                // tensor_dim0_stride[31:0] (data_size units)
      0, 0                             // stride hi / tensor_dim1_stride = 0
  };
  const i32x4 z4 = {0, 0, 0, 0};
#if defined(__clang_major__) && (__clang_major__ >= 23)
  const i32x8 z8 = {0, 0, 0, 0, 0, 0, 0, 0};
  __builtin_amdgcn_tensor_load_to_lds(g0, g1, z4, z4, z8, 0);
#else
  __builtin_amdgcn_tensor_load_to_lds(g0, g1, z4, z4, 0);
#endif
}
#endif  // HAVE_TDM

// -------------------- LayerNorm (f32 in -> bf16 out) --------------------
__global__ __launch_bounds__(256) void ln_bf16_kernel(const float* __restrict__ x,
                                                      const float* __restrict__ gamma,
                                                      const float* __restrict__ beta,
                                                      uint16_t* __restrict__ xn) {
  __shared__ float red[16];
  const int row = blockIdx.x;
  const int tid = threadIdx.x;               // 256 threads, 4 elems each
  const float4 v = reinterpret_cast<const float4*>(x + (size_t)row * DIMX)[tid];
  float s = v.x + v.y + v.z + v.w;
  float q = v.x * v.x + v.y * v.y + v.z * v.z + v.w * v.w;
  #pragma unroll
  for (int m = 1; m < 32; m <<= 1) { s += __shfl_xor(s, m, 32); q += __shfl_xor(q, m, 32); }
  const int wave = tid >> 5, lane = tid & 31;
  if (lane == 0) { red[wave] = s; red[8 + wave] = q; }
  __syncthreads();
  if (tid == 0) {
    float S = 0.f, Q = 0.f;
    #pragma unroll
    for (int i = 0; i < 8; ++i) { S += red[i]; Q += red[8 + i]; }
    red[0] = S; red[8] = Q;
  }
  __syncthreads();
  const float mu  = red[0] * (1.0f / DIMX);
  const float var = red[8] * (1.0f / DIMX) - mu * mu;
  const float rs  = rsqrtf(var + 1e-5f);
  const float4 g = reinterpret_cast<const float4*>(gamma)[tid];
  const float4 b = reinterpret_cast<const float4*>(beta)[tid];
  uint16_t* o = xn + (size_t)row * DIMX + tid * 4;
  o[0] = f2bf((v.x - mu) * rs * g.x + b.x);
  o[1] = f2bf((v.y - mu) * rs * g.y + b.y);
  o[2] = f2bf((v.z - mu) * rs * g.z + b.z);
  o[3] = f2bf((v.w - mu) * rs * g.w + b.w);
}

// -------------------- elementwise converts --------------------
__global__ __launch_bounds__(256) void cvt_bf16_kernel(const float* __restrict__ in,
                                                       uint16_t* __restrict__ out, int n) {
  int i = blockIdx.x * blockDim.x + threadIdx.x;
  if (i < n) out[i] = f2bf(in[i]);
}

// in: [K][N] f32 row-major  ->  out: [N][K] bf16 row-major (Bt layout for WMMA)
__global__ __launch_bounds__(256) void cvtT_bf16_kernel(const float* __restrict__ in,
                                                        uint16_t* __restrict__ out, int K, int N) {
  int i = blockIdx.x * blockDim.x + threadIdx.x;
  if (i < K * N) {
    int k = i / N, n = i % N;
    out[(size_t)n * K + k] = f2bf(in[i]);
  }
}

// kv: [B*NLAT][2*INNER] bf16 -> vT: [B][H][DHEAD][NLAT] bf16
__global__ __launch_bounds__(256) void extract_vT_kernel(const uint16_t* __restrict__ kv,
                                                         uint16_t* __restrict__ vT) {
  int i = blockIdx.x * blockDim.x + threadIdx.x;     // 2^21 elements
  int j = i & 1023;
  int d = (i >> 10) & 63;
  int h = (i >> 16) & 7;
  int b = i >> 19;
  vT[i] = kv[((size_t)(b * NLAT + j)) * (2 * INNER) + INNER + h * DHEAD + d];
}

// -------------------- 128x128x32 bf16 WMMA GEMM (TDM double-buffered) --------------------
// C[M,N] = scale * A[M,K] @ Bt[N,K]^T ; all dims multiples of 128/32.
template <bool OUTF32>
__global__ __launch_bounds__(256) void gemm_wmma_kernel(const uint16_t* __restrict__ A,
                                                        const uint16_t* __restrict__ Bt,
                                                        void* __restrict__ C,
                                                        int M, int N, int K, float scale) {
  (void)M;
  __shared__ uint16_t As[2][128 * 32];
  __shared__ uint16_t Bs[2][128 * 32];
  const int tid  = threadIdx.x;
  const int m0   = blockIdx.y * 128;
  const int n0   = blockIdx.x * 128;
  const int wave = tid >> 5;
  const int lane = tid & 31;
  const int half = lane >> 4;
  const int lm   = lane & 15;
  const int wm   = wave >> 1;        // 0..3 -> 32-row strip
  const int wn   = wave & 1;         // 0..1 -> 64-col strip

  f32x8 acc[2][4];
  #pragma unroll
  for (int i = 0; i < 2; ++i)
    #pragma unroll
    for (int j = 0; j < 4; ++j)
      #pragma unroll
      for (int r = 0; r < 8; ++r) acc[i][j][r] = 0.0f;

  const int nk = K / 32;
#if HAVE_TDM
  if (wave == 0) {   // prologue: DMA tile 0 into buffer 0
    tdm_load_tile_2d(lds_off(&As[0][0]), A + (size_t)m0 * K, 128, 32, K);
    tdm_load_tile_2d(lds_off(&Bs[0][0]), Bt + (size_t)n0 * K, 128, 32, K);
  }
#endif

  for (int it = 0; it < nk; ++it) {
    const int cur = it & 1;
    const int k0  = it * 32;
    __syncthreads();   // all waves done reading buffer !cur (previous iteration)
#if HAVE_TDM
    if (wave == 0) {
      if (it + 1 < nk) {   // overlap: DMA tile it+1 while computing tile it
        tdm_load_tile_2d(lds_off(&As[cur ^ 1][0]), A + (size_t)m0 * K + k0 + 32, 128, 32, K);
        tdm_load_tile_2d(lds_off(&Bs[cur ^ 1][0]), Bt + (size_t)n0 * K + k0 + 32, 128, 32, K);
        __builtin_amdgcn_s_wait_tensorcnt((short)2);   // tile it (2 loads) complete
      } else {
        __builtin_amdgcn_s_wait_tensorcnt((short)0);
      }
    }
#else
    {  // cooperative fallback copy of tile it into buffer cur
      const int lrow = tid >> 1;
      const int lseg = (tid & 1) * 16;
      const uint4* ga = reinterpret_cast<const uint4*>(A + (size_t)(m0 + lrow) * K + k0 + lseg);
      uint4* sa = reinterpret_cast<uint4*>(&As[cur][lrow * 32 + lseg]);
      sa[0] = ga[0]; sa[1] = ga[1];
      const uint4* gb = reinterpret_cast<const uint4*>(Bt + (size_t)(n0 + lrow) * K + k0 + lseg);
      uint4* sb = reinterpret_cast<uint4*>(&Bs[cur][lrow * 32 + lseg]);
      sb[0] = gb[0]; sb[1] = gb[1];
    }
#endif
    __syncthreads();   // tile it visible to all waves

    Frag a[2], b[4];
    #pragma unroll
    for (int i = 0; i < 2; ++i) {   // A 16x32 layout: lanes 0-15 K0-7/16-23, lanes 16-31 K8-15/24-31
      const uint16_t* p = &As[cur][(wm * 32 + i * 16 + lm) * 32 + half * 8];
      a[i].u[0] = *reinterpret_cast<const uint4*>(p);
      a[i].u[1] = *reinterpret_cast<const uint4*>(p + 16);
    }
    #pragma unroll
    for (int j = 0; j < 4; ++j) {   // B 32x16 layout: lane = column, lanes 0-15 K0-7/16-23, lanes 16-31 K8-15/24-31
      const uint16_t* p = &Bs[cur][(wn * 64 + j * 16 + lm) * 32 + half * 8];
      b[j].u[0] = *reinterpret_cast<const uint4*>(p);
      b[j].u[1] = *reinterpret_cast<const uint4*>(p + 16);
    }
    #pragma unroll
    for (int i = 0; i < 2; ++i)
      #pragma unroll
      for (int j = 0; j < 4; ++j)
        acc[i][j] = __builtin_amdgcn_wmma_f32_16x16x32_bf16(
            false, a[i].v, false, b[j].v, (short)0, acc[i][j], false, false);
  }

  #pragma unroll
  for (int i = 0; i < 2; ++i)
    #pragma unroll
    for (int j = 0; j < 4; ++j) {
      const int col = n0 + wn * 64 + j * 16 + lm;
      #pragma unroll
      for (int r = 0; r < 8; ++r) {   // C/D layout: VGPR r holds rows half*8+r
        const int row = m0 + wm * 32 + i * 16 + half * 8 + r;
        const float v = acc[i][j][r] * scale;
        if (OUTF32) reinterpret_cast<float*>(C)[(size_t)row * N + col] = v;
        else reinterpret_cast<uint16_t*>(C)[(size_t)row * N + col] = f2bf(v);
      }
    }
}

// -------------------- flash-attention (per (b,h), 128 q rows / block) --------------------
__global__ __launch_bounds__(256) void attn_kernel(const uint16_t* __restrict__ q,   // [B*NTOK][INNER]
                                                   const uint16_t* __restrict__ kv,  // [B*NLAT][2*INNER]
                                                   const uint16_t* __restrict__ vT,  // [B][H][DHEAD][NLAT]
                                                   const int* __restrict__ mask,     // [B][NLAT]
                                                   uint16_t* __restrict__ ao) {      // [B*NTOK][INNER]
  __shared__ uint16_t Qs[128 * 64];
  __shared__ uint16_t Ks[2][128 * 64];
  __shared__ uint16_t Vs[2][64 * 128];
  __shared__ uint16_t Ps[8 * 16 * 128];
  const int tid  = threadIdx.x;
  const int bh   = blockIdx.y;
  const int b    = bh >> 3;
  const int h    = bh & 7;
  const int m0   = blockIdx.x * 128;
  const int wave = tid >> 5, lane = tid & 31, half = lane >> 4, lm = lane & 15;

  const uint16_t* qbase = q + ((size_t)(b * NTOK + m0)) * INNER + h * DHEAD;
  const uint16_t* kbase = kv + ((size_t)(b * NLAT)) * (2 * INNER) + h * DHEAD;
  const uint16_t* vbase = vT + ((size_t)(b * HEADS + h)) * DHEAD * NLAT;

  __builtin_prefetch(mask + b * NLAT, 0, 3);   // global_prefetch: mask row stays hot

#if HAVE_TDM
  if (wave == 0) {   // prologue: Q once + first K/V tile
    tdm_load_tile_2d(lds_off(&Qs[0]),    qbase, 128, 64, INNER);
    tdm_load_tile_2d(lds_off(&Ks[0][0]), kbase, 128, 64, 2 * INNER);
    tdm_load_tile_2d(lds_off(&Vs[0][0]), vbase, 64, 128, NLAT);
  }
#else
  {
    const int row = tid >> 1, seg = (tid & 1) * 32;
    const uint4* g = reinterpret_cast<const uint4*>(qbase + (size_t)row * INNER + seg);
    uint4* s = reinterpret_cast<uint4*>(&Qs[row * 64 + seg]);
    s[0] = g[0]; s[1] = g[1]; s[2] = g[2]; s[3] = g[3];
  }
#endif

  float mrun[8], lsum[8];
  f32x8 o[4];
  #pragma unroll
  for (int r = 0; r < 8; ++r) { mrun[r] = NEG_BIG; lsum[r] = 0.0f; }
  #pragma unroll
  for (int dj = 0; dj < 4; ++dj)
    #pragma unroll
    for (int r = 0; r < 8; ++r) o[dj][r] = 0.0f;

  Frag aq[2];           // Q A-fragments: loop-invariant, loaded once in iter 0
  uint16_t* Pw = &Ps[wave * 16 * 128];
  const int njt = NLAT / 128;

  for (int it = 0; it < njt; ++it) {
    const int cur = it & 1;
    const int j0  = it * 128;
    __syncthreads();   // all waves done reading buffer !cur
#if HAVE_TDM
    if (wave == 0) {
      if (it + 1 < njt) {   // overlap next K/V DMA with this tile's compute
        tdm_load_tile_2d(lds_off(&Ks[cur ^ 1][0]), kbase + (size_t)(j0 + 128) * (2 * INNER),
                         128, 64, 2 * INNER);
        tdm_load_tile_2d(lds_off(&Vs[cur ^ 1][0]), vbase + (j0 + 128), 64, 128, NLAT);
        __builtin_amdgcn_s_wait_tensorcnt((short)2);
      } else {
        __builtin_amdgcn_s_wait_tensorcnt((short)0);
      }
    }
#else
    {
      const int row = tid >> 1, seg = (tid & 1) * 32;
      const uint4* gk = reinterpret_cast<const uint4*>(kbase + (size_t)(j0 + row) * (2 * INNER) + seg);
      uint4* sk = reinterpret_cast<uint4*>(&Ks[cur][row * 64 + seg]);
      sk[0] = gk[0]; sk[1] = gk[1]; sk[2] = gk[2]; sk[3] = gk[3];
      const int d = tid >> 2, seg2 = (tid & 3) * 32;
      const uint4* gv = reinterpret_cast<const uint4*>(vbase + (size_t)d * NLAT + j0 + seg2);
      uint4* sv = reinterpret_cast<uint4*>(&Vs[cur][d * 128 + seg2]);
      sv[0] = gv[0]; sv[1] = gv[1]; sv[2] = gv[2]; sv[3] = gv[3];
    }
#endif
    __syncthreads();

    if (it == 0) {   // Q tile guaranteed resident after first wait+barrier
      #pragma unroll
      for (int kk = 0; kk < 2; ++kk) {
        const uint16_t* ap = &Qs[(wave * 16 + lm) * 64 + kk * 32 + half * 8];
        aq[kk].u[0] = *reinterpret_cast<const uint4*>(ap);
        aq[kk].u[1] = *reinterpret_cast<const uint4*>(ap + 16);
      }
    }

    // sim tile: wave rows [wave*16, +16), all 128 latent cols; K-dim = DHEAD = 64
    f32x8 s[8];
    #pragma unroll
    for (int t = 0; t < 8; ++t)
      #pragma unroll
      for (int r = 0; r < 8; ++r) s[t][r] = 0.0f;
    #pragma unroll
    for (int kk = 0; kk < 2; ++kk) {
      #pragma unroll
      for (int t = 0; t < 8; ++t) {
        Frag bk;
        const uint16_t* bp = &Ks[cur][(t * 16 + lm) * 64 + kk * 32 + half * 8];
        bk.u[0] = *reinterpret_cast<const uint4*>(bp);
        bk.u[1] = *reinterpret_cast<const uint4*>(bp + 16);
        s[t] = __builtin_amdgcn_wmma_f32_16x16x32_bf16(
            false, aq[kk].v, false, bk.v, (short)0, s[t], false, false);
      }
    }

    // mask + online softmax (rows live in 16-lane halves -> shfl_xor 1..8)
    bool mk[8];
    #pragma unroll
    for (int t = 0; t < 8; ++t) mk[t] = mask[b * NLAT + j0 + t * 16 + lm] != 0;
    float rmax[8];
    #pragma unroll
    for (int r = 0; r < 8; ++r) rmax[r] = NEG_BIG;
    #pragma unroll
    for (int t = 0; t < 8; ++t)
      #pragma unroll
      for (int r = 0; r < 8; ++r) {
        float v = mk[t] ? s[t][r] : NEG_BIG;
        s[t][r] = v;
        rmax[r] = fmaxf(rmax[r], v);
      }
    #pragma unroll
    for (int m = 1; m < 16; m <<= 1)
      #pragma unroll
      for (int r = 0; r < 8; ++r) rmax[r] = fmaxf(rmax[r], __shfl_xor(rmax[r], m, 32));

    float mnew[8], alpha[8], rsum[8];
    #pragma unroll
    for (int r = 0; r < 8; ++r) {
      mnew[r]  = fmaxf(mrun[r], rmax[r]);
      alpha[r] = __expf(mrun[r] - mnew[r]);
      rsum[r]  = 0.0f;
    }
    #pragma unroll
    for (int t = 0; t < 8; ++t)
      #pragma unroll
      for (int r = 0; r < 8; ++r) {
        float p = __expf(s[t][r] - mnew[r]);
        rsum[r] += p;
        Pw[(half * 8 + r) * 128 + t * 16 + lm] = f2bf(p);   // D-layout -> row-major P in LDS
      }
    #pragma unroll
    for (int m = 1; m < 16; m <<= 1)
      #pragma unroll
      for (int r = 0; r < 8; ++r) rsum[r] += __shfl_xor(rsum[r], m, 32);
    #pragma unroll
    for (int r = 0; r < 8; ++r) {
      lsum[r] = lsum[r] * alpha[r] + rsum[r];
      mrun[r] = mnew[r];
    }
    #pragma unroll
    for (int dj = 0; dj < 4; ++dj)
      #pragma unroll
      for (int r = 0; r < 8; ++r) o[dj][r] *= alpha[r];

    __syncthreads();   // P visible before reload in A-layout

    // O += P[16x128] @ V[128x64] ; P reloaded in A-layout, Vs already in Bt layout
    #pragma unroll
    for (int kk = 0; kk < 128; kk += 32) {
      Frag pa;
      const uint16_t* pp = &Pw[lm * 128 + kk + half * 8];
      pa.u[0] = *reinterpret_cast<const uint4*>(pp);
      pa.u[1] = *reinterpret_cast<const uint4*>(pp + 16);
      #pragma unroll
      for (int dj = 0; dj < 4; ++dj) {
        Frag bv;
        const uint16_t* vp = &Vs[cur][(dj * 16 + lm) * 128 + kk + half * 8];
        bv.u[0] = *reinterpret_cast<const uint4*>(vp);
        bv.u[1] = *reinterpret_cast<const uint4*>(vp + 16);
        o[dj] = __builtin_amdgcn_wmma_f32_16x16x32_bf16(
            false, pa.v, false, bv.v, (short)0, o[dj], false, false);
      }
    }
  }

  #pragma unroll
  for (int dj = 0; dj < 4; ++dj)
    #pragma unroll
    for (int r = 0; r < 8; ++r) {
      const int row = m0 + wave * 16 + half * 8 + r;
      const int d   = dj * 16 + lm;
      float v = o[dj][r] / lsum[r];
      ao[((size_t)(b * NTOK + row)) * INNER + h * DHEAD + d] = f2bf(v);
    }
}

// -------------------- host launcher --------------------
extern "C" void kernel_launch(void* const* d_in, const int* in_sizes, int n_in,
                              void* d_out, int out_size, void* d_ws, size_t ws_size,
                              hipStream_t stream) {
  (void)in_sizes; (void)n_in; (void)out_size; (void)ws_size;
  const float* x     = (const float*)d_in[0];
  const float* media = (const float*)d_in[1];
  const int*   mask  = (const int*)d_in[2];
  const float* gamma = (const float*)d_in[3];
  const float* beta  = (const float*)d_in[4];
  const float* Wq    = (const float*)d_in[5];
  const float* Wkv   = (const float*)d_in[6];
  const float* Wout  = (const float*)d_in[7];
  float* out = (float*)d_out;

  uint16_t* p = (uint16_t*)d_ws;
  uint16_t* xn     = p; p += (size_t)BATCH * NTOK * DIMX;          // 32 MB
  uint16_t* mediab = p; p += (size_t)BATCH * NLAT * DIMX;          //  8 MB
  uint16_t* WqT    = p; p += (size_t)INNER * DIMX;                 //  1 MB
  uint16_t* WkvT   = p; p += (size_t)(2 * INNER) * DIMX;           //  2 MB
  uint16_t* WoutT  = p; p += (size_t)DIMX * INNER;                 //  1 MB
  uint16_t* qb     = p; p += (size_t)BATCH * NTOK * INNER;         // 16 MB
  uint16_t* kvb    = p; p += (size_t)BATCH * NLAT * 2 * INNER;     //  8 MB
  uint16_t* vT     = p; p += (size_t)BATCH * HEADS * DHEAD * NLAT; //  4 MB
  uint16_t* ao     = p; p += (size_t)BATCH * NTOK * INNER;         // 16 MB

  ln_bf16_kernel<<<BATCH * NTOK, 256, 0, stream>>>(x, gamma, beta, xn);
  cvt_bf16_kernel<<<(BATCH * NLAT * DIMX) / 256, 256, 0, stream>>>(media, mediab, BATCH * NLAT * DIMX);
  cvtT_bf16_kernel<<<(DIMX * INNER) / 256, 256, 0, stream>>>(Wq, WqT, DIMX, INNER);
  cvtT_bf16_kernel<<<(DIMX * 2 * INNER) / 256, 256, 0, stream>>>(Wkv, WkvT, DIMX, 2 * INNER);
  cvtT_bf16_kernel<<<(INNER * DIMX) / 256, 256, 0, stream>>>(Wout, WoutT, INNER, DIMX);

  const float scale = 0.125f;  // DHEAD^-0.5
  gemm_wmma_kernel<false><<<dim3(INNER / 128, (BATCH * NTOK) / 128), 256, 0, stream>>>(
      xn, WqT, qb, BATCH * NTOK, INNER, DIMX, scale);
  gemm_wmma_kernel<false><<<dim3((2 * INNER) / 128, (BATCH * NLAT) / 128), 256, 0, stream>>>(
      mediab, WkvT, kvb, BATCH * NLAT, 2 * INNER, DIMX, 1.0f);
  extract_vT_kernel<<<(BATCH * HEADS * DHEAD * NLAT) / 256, 256, 0, stream>>>(kvb, vT);
  attn_kernel<<<dim3(NTOK / 128, BATCH * HEADS), 256, 0, stream>>>(qb, kvb, vT, mask, ao);
  gemm_wmma_kernel<true><<<dim3(DIMX / 128, (BATCH * NTOK) / 128), 256, 0, stream>>>(
      ao, WoutT, out, BATCH * NTOK, DIMX, INNER, 1.0f);
}
